// YOLOv8Head_64407329570837
// MI455X (gfx1250) — compile-verified, matrix operand order checked by reading, not run on hardware
//
#include <hip/hip_runtime.h>
#include <hip/hip_bf16.h>

typedef __attribute__((ext_vector_type(16))) __bf16 v16bf;
typedef __attribute__((ext_vector_type(8)))  __bf16 bf16x8;
typedef __attribute__((ext_vector_type(8)))  float  v8f;

// GCC-style vector type matching the async-LDS builtin's expected param type:
//   '__attribute__((__vector_size__(4 * sizeof(int)))) int __device__ *'
typedef int i32x4_g __attribute__((vector_size(16)));
typedef __attribute__((address_space(1))) i32x4_g* gas_b128_ptr;
typedef __attribute__((address_space(3))) i32x4_g* las_b128_ptr;

#if defined(__has_builtin)
# if __has_builtin(__builtin_amdgcn_global_load_async_to_lds_b128)
#  define HAVE_ASYNC_LDS 1
# endif
#endif
#ifndef HAVE_ASYNC_LDS
# define HAVE_ASYNC_LDS 0
#endif

__device__ __forceinline__ void wait_async_then_barrier() {
#if HAVE_ASYNC_LDS
# if defined(__has_builtin) && __has_builtin(__builtin_amdgcn_s_wait_asynccnt)
    __builtin_amdgcn_s_wait_asynccnt(0);
# else
    asm volatile("s_wait_asynccnt 0" ::: "memory");
# endif
#endif
    __syncthreads();
}

// ---------- fp32 -> bf16 (grid-stride) ----------
__global__ void cvt_f32_to_bf16(const float* __restrict__ src,
                                __bf16* __restrict__ dst, int n) {
    for (int i = blockIdx.x * blockDim.x + threadIdx.x; i < n;
         i += gridDim.x * blockDim.x)
        dst[i] = (__bf16)src[i];
}

// ---------- weight prep: OIHW (C,C,3,3) fp32 -> [tap][o][i] bf16 ----------
__global__ void prep_weights(const float* __restrict__ w,
                             __bf16* __restrict__ wt, int C) {
    int n = C * C * 9;
    for (int idx = blockIdx.x * blockDim.x + threadIdx.x; idx < n;
         idx += gridDim.x * blockDim.x) {
        int t = idx / (C * C);
        int r = idx - t * C * C;
        int o = r / C;
        int i = r - o * C;
        wt[idx] = (__bf16)w[(o * C + i) * 9 + t];
    }
}

// ---------- fused conv3x3 (pad 1) + BN scale/bias + ReLU, implicit GEMM ----------
// Block: 256 threads = 8 waves. Tile: 64 couts x 64 spatial.
// Wave (mw,nw): mw in 0..3 -> cout 16-block, nw in 0..1 -> spatial 32-block.
// Double-buffered LDS, one barrier per K-step, 2 WMMAs per wave per K-step.
// A tile staged memory->LDS via GLOBAL_LOAD_ASYNC_TO_LDS_B128 when available.
__global__ __launch_bounds__(256)
void conv3x3_bn_relu_wmma(const __bf16* __restrict__ x,
                          const __bf16* __restrict__ wt,   // [tap][o][i] bf16
                          const float* __restrict__ scale,
                          const float* __restrict__ bias,
                          __bf16* __restrict__ y,
                          int C, int H, int W) {
    __shared__ __align__(16) __bf16 As[2][64 * 32];   // [buf][cout][k]  2x4KB
    __shared__ __align__(16) __bf16 Bs[2][64 * 32];   // [buf][pos][k]   2x4KB

    const int tid  = threadIdx.x;
    const int lane = tid & 31;
    const int wv   = tid >> 5;
    const int mw   = wv & 3;       // cout 16-block
    const int nw   = wv >> 2;      // spatial 32-block
    const int HW   = H * W;
    const int pBase    = blockIdx.x * 64;
    const int coutBase = blockIdx.y * 64;
    const __bf16* xb = x + (size_t)blockIdx.z * C * HW;

    // A staging: 8 contiguous bf16 (16B) per thread, covers 64x32
    const int aIdx = tid * 8;
    const int aRow = aIdx >> 5;
    const int aCol = aIdx & 31;
    // B staging: one spatial position, 8 consecutive channels per thread
    const int pLoc = tid >> 2;             // 0..63
    const int c8   = (tid & 3) * 8;
    const int p    = pBase + pLoc;
    const int py   = p / W;
    const int px   = p - py * W;

    const int csteps = C >> 5;             // channel chunks of 32
    const int ksteps = 9 * csteps;

    v8f acc0 = {}, acc1 = {};

    auto stage = [&](int ks, int buf) {
        const int t  = ks / csteps;
        const int c0 = (ks - t * csteps) << 5;
        const int dy = t / 3 - 1;
        const int dx = t - (t / 3) * 3 - 1;

        // ---- A tile (64 couts x 32 k): contiguous 16B per thread ----
        const __bf16* asrc =
            wt + ((size_t)t * C + (coutBase + aRow)) * C + (c0 + aCol);
#if HAVE_ASYNC_LDS
        __builtin_amdgcn_global_load_async_to_lds_b128(
            (gas_b128_ptr)(uintptr_t)asrc,
            (las_b128_ptr)(uint32_t)(uintptr_t)&As[buf][aIdx],
            0, 0);
#else
        *(int4*)(&As[buf][aIdx]) = *(const int4*)asrc;
        __builtin_prefetch(asrc + 32, 0, 1);   // global_prefetch_b8
#endif

        // ---- B tile (64 pos x 32 ch), im2col gather with zero pad ----
        const int yy = py + dy;
        const int xx = px + dx;
        bf16x8 bv = {};
        if ((yy >= 0) && (yy < H) && (xx >= 0) && (xx < W)) {
            const __bf16* bsrc = xb + (size_t)(c0 + c8) * HW + yy * W + xx;
#pragma unroll
            for (int j = 0; j < 8; ++j)
                bv[j] = bsrc[(size_t)j * HW];
        }
        *(bf16x8*)(&Bs[buf][pLoc * 32 + c8]) = bv;
    };

    stage(0, 0);
    wait_async_then_barrier();

    for (int ks = 0; ks < ksteps; ++ks) {
        const int cur = ks & 1;
        if (ks + 1 < ksteps)
            stage(ks + 1, cur ^ 1);

        // ---- fragment loads per ISA 16-bit layout ----
        // lanes 0-15: K 0..7 / 16..23 ; lanes 16-31: K 8..15 / 24..31
        const int kb   = (lane >> 4) * 8;
        const int mRow = mw * 16 + (lane & 15);
        const int nCol = nw * 32 + (lane & 15);

        bf16x8 a0 = *(const bf16x8*)(&As[cur][mRow * 32 + kb]);
        bf16x8 a1 = *(const bf16x8*)(&As[cur][mRow * 32 + kb + 16]);
        v16bf Af = __builtin_shufflevector(a0, a1, 0,1,2,3,4,5,6,7,
                                                   8,9,10,11,12,13,14,15);

        bf16x8 b0 = *(const bf16x8*)(&Bs[cur][nCol * 32 + kb]);
        bf16x8 b1 = *(const bf16x8*)(&Bs[cur][nCol * 32 + kb + 16]);
        v16bf B0 = __builtin_shufflevector(b0, b1, 0,1,2,3,4,5,6,7,
                                                   8,9,10,11,12,13,14,15);
        bf16x8 b2 = *(const bf16x8*)(&Bs[cur][(nCol + 16) * 32 + kb]);
        bf16x8 b3 = *(const bf16x8*)(&Bs[cur][(nCol + 16) * 32 + kb + 16]);
        v16bf B1 = __builtin_shufflevector(b2, b3, 0,1,2,3,4,5,6,7,
                                                   8,9,10,11,12,13,14,15);

        acc0 = __builtin_amdgcn_wmma_f32_16x16x32_bf16(
            false, Af, false, B0, (short)0, acc0, false, false);
        acc1 = __builtin_amdgcn_wmma_f32_16x16x32_bf16(
            false, Af, false, B1, (short)0, acc1, false, false);

        wait_async_then_barrier();
    }

    // ---- epilogue: BN scale/bias + ReLU, bf16 store ----
    // C/D layout: VGPR r -> M = r + (lane>=16 ? 8 : 0), N = lane%16
    const int nOut = pBase + nw * 32 + (lane & 15);
    const int mOff = coutBase + mw * 16 + ((lane >> 4) << 3);
    __bf16* yb = y + (size_t)blockIdx.z * C * HW;
#pragma unroll
    for (int r = 0; r < 8; ++r) {
        const int m = mOff + r;
        const float s = scale[m];
        const float bb = bias[m];
        float v0 = acc0[r] * s + bb; v0 = v0 > 0.0f ? v0 : 0.0f;
        float v1 = acc1[r] * s + bb; v1 = v1 > 0.0f ? v1 : 0.0f;
        yb[(size_t)m * HW + nOut]      = (__bf16)v0;
        yb[(size_t)m * HW + nOut + 16] = (__bf16)v1;
    }
}

// ---------- 1x1 conv (C->5) + bias + anchor decode + sigmoid ----------
__global__ void head_final_decode(
    const __bf16* __restrict__ y3, const __bf16* __restrict__ y4,
    const __bf16* __restrict__ y5,
    const float* __restrict__ wf3, const float* __restrict__ bf3,
    const float* __restrict__ wf4, const float* __restrict__ bf4,
    const float* __restrict__ wf5, const float* __restrict__ bf5,
    float* __restrict__ out) {
    const int TOT = 33600;           // 160^2 + 80^2 + 40^2
    const int a = blockIdx.x * blockDim.x + threadIdx.x;
    const int b = blockIdx.y;
    if (a >= TOT) return;

    const __bf16* yb; const float* wf; const float* bfv;
    int C, Wl, p; float strd;
    if (a < 25600)      { yb = y3; wf = wf3; bfv = bf3; C = 128; Wl = 160; strd = 8.f;  p = a; }
    else if (a < 32000) { yb = y4; wf = wf4; bfv = bf4; C = 256; Wl = 80;  strd = 16.f; p = a - 25600; }
    else                { yb = y5; wf = wf5; bfv = bf5; C = 512; Wl = 40;  strd = 32.f; p = a - 32000; }

    const int HW = Wl * Wl;
    const __bf16* src = yb + (size_t)b * C * HW + p;
    float a0 = 0.f, a1 = 0.f, a2 = 0.f, a3 = 0.f, a4 = 0.f;
#pragma unroll 4
    for (int i = 0; i < C; ++i) {
        float xv = (float)src[(size_t)i * HW];
        a0 += xv * wf[i];
        a1 += xv * wf[C + i];
        a2 += xv * wf[2 * C + i];
        a3 += xv * wf[3 * C + i];
        a4 += xv * wf[4 * C + i];
    }
    a0 += bfv[0]; a1 += bfv[1]; a2 += bfv[2]; a3 += bfv[3]; a4 += bfv[4];

    const float ax = (float)(p % Wl) + 0.5f;
    const float ay = (float)(p / Wl) + 0.5f;
    const float x1 = ax - a0, yl = ay - a1;
    const float x2 = ax + a2, y2v = ay + a3;
    const float cx = (x1 + x2) * 0.5f * strd;
    const float cy = (yl + y2v) * 0.5f * strd;
    const float bw = (x2 - x1) * strd;
    const float bh = (y2v - yl) * strd;
    const float cls = 1.0f / (1.0f + __expf(-a4));

    const size_t ob = (size_t)b * 5 * TOT;
    out[ob + 0 * TOT + a] = cx;
    out[ob + 1 * TOT + a] = cy;
    out[ob + 2 * TOT + a] = bw;
    out[ob + 3 * TOT + a] = bh;
    out[ob + 4 * TOT + a] = cls;
}

extern "C" void kernel_launch(void* const* d_in, const int* in_sizes, int n_in,
                              void* d_out, int out_size, void* d_ws, size_t ws_size,
                              hipStream_t stream) {
    (void)in_sizes; (void)n_in; (void)out_size; (void)ws_size;
    const int B = 8;
    const int Cs[3]  = {128, 256, 512};
    const int HWs[3] = {160, 80, 40};
    const float* xin[3] = {(const float*)d_in[0], (const float*)d_in[1],
                           (const float*)d_in[2]};

    size_t S[3], Stot = 0;
    for (int l = 0; l < 3; ++l) {
        S[l] = (size_t)B * Cs[l] * HWs[l] * HWs[l];
        Stot += S[l];
    }

    __bf16* ws  = (__bf16*)d_ws;
    __bf16* xb  = ws;              // bf16 inputs; reused as conv2 outputs
    __bf16* y1  = ws + Stot;       // conv1 outputs
    __bf16* wtb = ws + 2 * Stot;   // transformed weights

    size_t xoff[3] = {0, S[0], S[0] + S[1]};
    __bf16* wt1[3]; __bf16* wt2[3];
    {
        size_t woff = 0;
        for (int l = 0; l < 3; ++l) {
            size_t wsz = (size_t)Cs[l] * Cs[l] * 9;
            wt1[l] = wtb + woff; woff += wsz;
            wt2[l] = wtb + woff; woff += wsz;
        }
    }

    for (int l = 0; l < 3; ++l) {
        const int C = Cs[l], Hh = HWs[l], Ww = HWs[l];
        const int base = 3 + 8 * l;
        const float* w1 = (const float*)d_in[base + 0];
        const float* s1 = (const float*)d_in[base + 1];
        const float* b1 = (const float*)d_in[base + 2];
        const float* w2 = (const float*)d_in[base + 3];
        const float* s2 = (const float*)d_in[base + 4];
        const float* b2 = (const float*)d_in[base + 5];

        cvt_f32_to_bf16<<<dim3(2048), dim3(256), 0, stream>>>(
            xin[l], xb + xoff[l], (int)S[l]);

        int wn = C * C * 9;
        prep_weights<<<dim3((wn + 255) / 256), dim3(256), 0, stream>>>(w1, wt1[l], C);
        prep_weights<<<dim3((wn + 255) / 256), dim3(256), 0, stream>>>(w2, wt2[l], C);

        dim3 grid(Hh * Ww / 64, C / 64, B);
        conv3x3_bn_relu_wmma<<<grid, dim3(256), 0, stream>>>(
            xb + xoff[l], wt1[l], s1, b1, y1 + xoff[l], C, Hh, Ww);
        conv3x3_bn_relu_wmma<<<grid, dim3(256), 0, stream>>>(
            y1 + xoff[l], wt2[l], s2, b2, xb + xoff[l], C, Hh, Ww);
    }

    head_final_decode<<<dim3((33600 + 255) / 256, B), dim3(256), 0, stream>>>(
        xb + xoff[0], xb + xoff[1], xb + xoff[2],
        (const float*)d_in[9],  (const float*)d_in[10],
        (const float*)d_in[17], (const float*)d_in[18],
        (const float*)d_in[25], (const float*)d_in[26],
        (float*)d_out);
}
